// MagNet_43061342109778
// MI455X (gfx1250) — matrix operand order repaired and checked by reading, not compiled, not run on hardware
//
#include <hip/hip_runtime.h>
#include <stdint.h>

// ---------------------------------------------------------------------------
// MagNet forward for MI455X (gfx1250): bf16 WMMA path.
// B=32, N=512, F=AGG=128, FCH=256, NCLS=10, NLAYER=3
// ---------------------------------------------------------------------------

typedef __attribute__((ext_vector_type(16))) __bf16 v16bf;
typedef __attribute__((ext_vector_type(8)))  float  v8f;
typedef unsigned short u16t;

__device__ __forceinline__ u16t f2bf(float f) {
  unsigned int u = __float_as_uint(f);
  unsigned int r = u + 0x7FFFu + ((u >> 16) & 1u);   // round-to-nearest-even
  return (u16t)(r >> 16);
}
__device__ __forceinline__ float bf2f(u16t h) {
  return __uint_as_float(((unsigned int)h) << 16);
}

__device__ __forceinline__ v8f wmma_bf16(v16bf a, v16bf b, v8f c) {
  // D = A(16x32 bf16) * B(32x16 bf16) + C(16x16 f32)
  return __builtin_amdgcn_wmma_f32_16x16x32_bf16(false, a, false, b,
                                                 (short)0, c, false, false);
}

// load a 16-element bf16 fragment as two 16B chunks at p0 and p1
__device__ __forceinline__ v16bf load_frag(const u16t* p0, const u16t* p1) {
  v16bf f;
  ((uint4*)&f)[0] = *(const uint4*)p0;
  ((uint4*)&f)[1] = *(const uint4*)p1;
  return f;
}

// ---------------------------------------------------------------------------
// Setup: f32 -> bf16 conversions
// ---------------------------------------------------------------------------

// L_re/L_im [32,512,512] f32 -> bf16 row-major. 4 elems/thread, exact grid.
__global__ void conv_L(const float* __restrict__ re, const float* __restrict__ im,
                       u16t* __restrict__ re16, u16t* __restrict__ im16) {
  const size_t i = ((size_t)blockIdx.x * blockDim.x + threadIdx.x) * 4;
  float4 a = *(const float4*)(re + i);
  float4 c = *(const float4*)(im + i);
  ushort4 oa, oc;
  oa.x = f2bf(a.x); oa.y = f2bf(a.y); oa.z = f2bf(a.z); oa.w = f2bf(a.w);
  oc.x = f2bf(c.x); oc.y = f2bf(c.y); oc.z = f2bf(c.z); oc.w = f2bf(c.w);
  *(ushort4*)(re16 + i) = oa;
  *(ushort4*)(im16 + i) = oc;
}

// x [32,512,128] f32 -> hT[b][f][node] bf16 (transposed), him = 0.
__global__ void conv_x(const float* __restrict__ x,
                       u16t* __restrict__ hre, u16t* __restrict__ him) {
  const int i = blockIdx.x * blockDim.x + threadIdx.x;   // 0 .. 32*128*512-1
  const int n = i & 511;
  const int f = (i >> 9) & 127;
  const int b = i >> 16;
  hre[i] = f2bf(x[(((size_t)b << 9) + n) * 128 + f]);
  him[i] = 0;
}

// Wr/Wi [3,128,128] (k,o) -> WT[l][o][k] bf16 (transposed per layer).
__global__ void conv_w(const float* __restrict__ Wr, const float* __restrict__ Wi,
                       u16t* __restrict__ tre, u16t* __restrict__ tim) {
  const int i = blockIdx.x * blockDim.x + threadIdx.x;   // 0 .. 3*128*128-1
  const int k = i & 127;
  const int o = (i >> 7) & 127;
  const int l = i >> 14;
  const size_t src = (((size_t)l * 128 + k) * 128) + o;
  tre[i] = f2bf(Wr[src]);
  tim[i] = f2bf(Wi[src]);
}

// ---------------------------------------------------------------------------
// Aggregation: agg = L @ h  (complex), per batch [512x512]@[512x128].
// A = L (row-major bf16), B = hT[b][f][node] (transposed -> contiguous cols).
// Each wave: one 32x32 output tile (2x2 WMMA tiles), K-loop over 512.
// Writes agg row-major bf16 [b][node][f].
// ---------------------------------------------------------------------------
__global__ __launch_bounds__(128, 1) void agg_kernel(
    const u16t* __restrict__ Lre, const u16t* __restrict__ Lim,
    const u16t* __restrict__ hTre, const u16t* __restrict__ hTim,
    u16t* __restrict__ aggre, u16t* __restrict__ aggim) {
  const int lane = threadIdx.x & 31;
  const int wave = threadIdx.x >> 5;
  const int tile = blockIdx.x * 4 + wave;      // 0..2047
  const int b  = tile >> 6;                    // 64 tiles per batch
  const int tt = tile & 63;
  const int m0 = (tt >> 2) << 5;               // node tile (16 of them)
  const int n0 = (tt & 3) << 5;                // feature tile (4 of them)
  const int hk = lane >> 4;                    // half: 0/1
  const int lm = lane & 15;

  const u16t* Lre_b = Lre + ((size_t)b << 18);
  const u16t* Lim_b = Lim + ((size_t)b << 18);
  const u16t* hre_b = hTre + ((size_t)b << 16);
  const u16t* him_b = hTim + ((size_t)b << 16);

  v8f acr[2][2] = {};
  v8f aci[2][2] = {};

  for (int kk = 0; kk < 512; kk += 32) {
    v16bf Are[2], Aim[2];
#pragma unroll
    for (int mh = 0; mh < 2; ++mh) {
      const int M = m0 + (mh << 4) + lm;
      const u16t* pr = Lre_b + (((size_t)M) << 9) + kk + (hk << 3);
      const u16t* pi = Lim_b + (((size_t)M) << 9) + kk + (hk << 3);
      Are[mh] = load_frag(pr, pr + 16);        // K = base+{0..7}, base+16+{0..7}
      Aim[mh] = load_frag(pi, pi + 16);
    }
    v16bf Bre[2], Bim[2], Bni[2];
#pragma unroll
    for (int nh = 0; nh < 2; ++nh) {
      const int Nc = n0 + (nh << 4) + lm;
      const u16t* pr = hre_b + (((size_t)Nc) << 9) + kk + (hk << 3);
      const u16t* pi = him_b + (((size_t)Nc) << 9) + kk + (hk << 3);
      Bre[nh] = load_frag(pr, pr + 16);        // K = base+{0..7}, base+16+{0..7}
      Bim[nh] = load_frag(pi, pi + 16);
#pragma unroll
      for (int j = 0; j < 8; ++j)              // -him via bf16 sign flip
        ((unsigned int*)&Bni[nh])[j] =
            ((const unsigned int*)&Bim[nh])[j] ^ 0x80008000u;
    }
#pragma unroll
    for (int mh = 0; mh < 2; ++mh)
#pragma unroll
      for (int nh = 0; nh < 2; ++nh) {
        acr[mh][nh] = wmma_bf16(Are[mh], Bre[nh], acr[mh][nh]);
        acr[mh][nh] = wmma_bf16(Aim[mh], Bni[nh], acr[mh][nh]);  // re -= Lim*him
        aci[mh][nh] = wmma_bf16(Are[mh], Bim[nh], aci[mh][nh]);
        aci[mh][nh] = wmma_bf16(Aim[mh], Bre[nh], aci[mh][nh]);
      }
  }

#pragma unroll
  for (int mh = 0; mh < 2; ++mh)
#pragma unroll
    for (int nh = 0; nh < 2; ++nh)
#pragma unroll
      for (int r = 0; r < 8; ++r) {
        const int M  = m0 + (mh << 4) + (hk << 3) + r;   // C layout: M=r+8*half
        const int Nc = n0 + (nh << 4) + lm;              // N=lane%16
        const size_t o = (((size_t)b << 9) + M) * 128 + Nc;
        aggre[o] = f2bf(acr[mh][nh][r]);
        aggim[o] = f2bf(aci[mh][nh][r]);
      }
}

// ---------------------------------------------------------------------------
// Transform + complex ReLU: t = agg @ W, h_next = (Re(t)>=0) ? t : 0.
// A = agg row-major [b][node][128]; B = WT[o][k] (transposed, per layer).
// Output stored TRANSPOSED hT[b][f][node] for the next aggregation / reduce.
// ---------------------------------------------------------------------------
__global__ __launch_bounds__(128, 1) void xform_kernel(
    const u16t* __restrict__ aggre, const u16t* __restrict__ aggim,
    const u16t* __restrict__ WTre, const u16t* __restrict__ WTim,
    u16t* __restrict__ hTre, u16t* __restrict__ hTim) {
  const int lane = threadIdx.x & 31;
  const int wave = threadIdx.x >> 5;
  const int tile = blockIdx.x * 4 + wave;      // 0..2047
  const int b  = tile >> 6;
  const int tt = tile & 63;
  const int m0 = (tt >> 2) << 5;               // node tile
  const int n0 = (tt & 3) << 5;                // out-feature tile
  const int hk = lane >> 4;
  const int lm = lane & 15;

  const u16t* are_b = aggre + ((size_t)b << 16);
  const u16t* aim_b = aggim + ((size_t)b << 16);

  v8f acr[2][2] = {};
  v8f aci[2][2] = {};

  for (int kk = 0; kk < 128; kk += 32) {
    v16bf Are[2], Aim[2];
#pragma unroll
    for (int mh = 0; mh < 2; ++mh) {
      const int M = m0 + (mh << 4) + lm;
      const u16t* pr = are_b + (((size_t)M) << 7) + kk + (hk << 3);
      const u16t* pi = aim_b + (((size_t)M) << 7) + kk + (hk << 3);
      Are[mh] = load_frag(pr, pr + 16);
      Aim[mh] = load_frag(pi, pi + 16);
    }
    v16bf Bre[2], Bim[2], Bni[2];
#pragma unroll
    for (int nh = 0; nh < 2; ++nh) {
      const int Nc = n0 + (nh << 4) + lm;
      const u16t* pr = WTre + (((size_t)Nc) << 7) + kk + (hk << 3);
      const u16t* pi = WTim + (((size_t)Nc) << 7) + kk + (hk << 3);
      Bre[nh] = load_frag(pr, pr + 16);
      Bim[nh] = load_frag(pi, pi + 16);
#pragma unroll
      for (int j = 0; j < 8; ++j)
        ((unsigned int*)&Bni[nh])[j] =
            ((const unsigned int*)&Bim[nh])[j] ^ 0x80008000u;
    }
#pragma unroll
    for (int mh = 0; mh < 2; ++mh)
#pragma unroll
      for (int nh = 0; nh < 2; ++nh) {
        acr[mh][nh] = wmma_bf16(Are[mh], Bre[nh], acr[mh][nh]);
        acr[mh][nh] = wmma_bf16(Aim[mh], Bni[nh], acr[mh][nh]);
        aci[mh][nh] = wmma_bf16(Are[mh], Bim[nh], aci[mh][nh]);
        aci[mh][nh] = wmma_bf16(Aim[mh], Bre[nh], aci[mh][nh]);
      }
  }

#pragma unroll
  for (int mh = 0; mh < 2; ++mh)
#pragma unroll
    for (int nh = 0; nh < 2; ++nh)
#pragma unroll
      for (int r = 0; r < 8; ++r) {
        const int M  = m0 + (mh << 4) + (hk << 3) + r;
        const int Nc = n0 + (nh << 4) + lm;
        float re = acr[mh][nh][r];
        float im = aci[mh][nh][r];
        if (re < 0.0f) { re = 0.0f; im = 0.0f; }           // complex ReLU
        const size_t o = (((size_t)b << 7) + Nc) * 512 + M; // transposed store
        hTre[o] = f2bf(re);
        hTim[o] = f2bf(im);
      }
}

// ---------------------------------------------------------------------------
// Readout: feat[b][c] = mean over nodes; c<128 -> Re, c>=128 -> Im.
// hT layout makes each sum a contiguous 512-element run.
// ---------------------------------------------------------------------------
__global__ void reduce_mean(const u16t* __restrict__ hTre,
                            const u16t* __restrict__ hTim,
                            float* __restrict__ feat) {
  const int i = blockIdx.x * blockDim.x + threadIdx.x;  // 0..8191
  const int c = i & 255;
  const int b = i >> 8;
  const u16t* src = (c < 128) ? hTre : hTim;
  const int f = c & 127;
  const u16t* p = src + (((size_t)b << 7) + f) * 512;
  float s = 0.0f;
  for (int m = 0; m < 512; ++m) s += bf2f(p[m]);
  feat[i] = s * (1.0f / 512.0f);
}

// ---------------------------------------------------------------------------
// FC head + softmax, one block per batch row (f32).
// ---------------------------------------------------------------------------
__global__ __launch_bounds__(256) void head_kernel(
    const float* __restrict__ feat,
    const float* __restrict__ w1, const float* __restrict__ b1,
    const float* __restrict__ w2, const float* __restrict__ b2,
    float* __restrict__ out) {
  __shared__ float zs[256];
  __shared__ float lg[10];
  const int b = blockIdx.x;
  const int t = threadIdx.x;
  const float* fv = feat + b * 256;

  float acc = b1[t];
  const float* wr = w1 + t * 256;
  for (int k = 0; k < 256; ++k) acc += fv[k] * wr[k];
  zs[t] = fmaxf(acc, 0.0f);
  __syncthreads();

  if (t < 10) {
    float a = b2[t];
    const float* w = w2 + t * 256;
    for (int k = 0; k < 256; ++k) a += zs[k] * w[k];
    lg[t] = a;
  }
  __syncthreads();

  if (t == 0) {
    float mx = lg[0];
    for (int j = 1; j < 10; ++j) mx = fmaxf(mx, lg[j]);
    float e[10], s = 0.0f;
    for (int j = 0; j < 10; ++j) { e[j] = expf(lg[j] - mx); s += e[j]; }
    const float inv = 1.0f / s;
    for (int j = 0; j < 10; ++j) out[b * 10 + j] = e[j] * inv;
  }
}

// ---------------------------------------------------------------------------
extern "C" void kernel_launch(void* const* d_in, const int* in_sizes, int n_in,
                              void* d_out, int out_size, void* d_ws, size_t ws_size,
                              hipStream_t stream) {
  (void)in_sizes; (void)n_in; (void)out_size; (void)ws_size;
  const float* x    = (const float*)d_in[0];
  const float* L_re = (const float*)d_in[1];
  const float* L_im = (const float*)d_in[2];
  const float* Wr   = (const float*)d_in[3];
  const float* Wi   = (const float*)d_in[4];
  const float* w1   = (const float*)d_in[5];
  const float* b1   = (const float*)d_in[6];
  const float* w2   = (const float*)d_in[7];
  const float* b2   = (const float*)d_in[8];
  float* out = (float*)d_out;

  char* ws = (char*)d_ws;
  size_t off = 0;
  auto nxt = [&](size_t bytes) { void* p = ws + off; off += bytes; return p; };

  u16t* Lre16 = (u16t*)nxt(16777216);   // 32*512*512 bf16
  u16t* Lim16 = (u16t*)nxt(16777216);
  u16t* h0re  = (u16t*)nxt(4194304);    // 32*128*512 bf16 (hT)
  u16t* h0im  = (u16t*)nxt(4194304);
  u16t* h1re  = (u16t*)nxt(4194304);
  u16t* h1im  = (u16t*)nxt(4194304);
  u16t* agre  = (u16t*)nxt(4194304);    // 32*512*128 bf16
  u16t* agim  = (u16t*)nxt(4194304);
  u16t* WTre  = (u16t*)nxt(98304);      // 3*128*128 bf16
  u16t* WTim  = (u16t*)nxt(98304);
  float* feat = (float*)nxt(32768);     // 32*256 f32

  conv_L<<<8192, 256, 0, stream>>>(L_re, L_im, Lre16, Lim16);
  conv_x<<<8192, 256, 0, stream>>>(x, h0re, h0im);
  conv_w<<<192, 256, 0, stream>>>(Wr, Wi, WTre, WTim);

  u16t* cre = h0re; u16t* cim = h0im;
  u16t* nre = h1re; u16t* nim = h1im;
  for (int l = 0; l < 3; ++l) {
    agg_kernel<<<512, 128, 0, stream>>>(Lre16, Lim16, cre, cim, agre, agim);
    xform_kernel<<<512, 128, 0, stream>>>(agre, agim,
                                          WTre + l * 16384, WTim + l * 16384,
                                          nre, nim);
    u16t* t;
    t = cre; cre = nre; nre = t;
    t = cim; cim = nim; nim = t;
  }

  reduce_mean<<<32, 256, 0, stream>>>(cre, cim, feat);
  head_kernel<<<32, 256, 0, stream>>>(feat, w1, b1, w2, b2, out);
}
